// OGBMOLModel3_16956530884983
// MI455X (gfx1250) — compile-verified, matrix operand order checked by reading, not run on hardware
//
#include <hip/hip_runtime.h>
#include <hip/hip_bf16.h>

// ---------------------------------------------------------------------------
// DR2FWL2-style GNN for MI455X (gfx1250, wave32).
// Dense 128x128 linears run on the bf16 WMMA matrix pipe
// (v_wmma_f32_16x16x32_bf16), everything else is coalesced row kernels with
// f32 global atomics for the scatter-adds.
// ---------------------------------------------------------------------------

typedef __attribute__((ext_vector_type(16))) __bf16 v16bf;
typedef __attribute__((ext_vector_type(8)))  float  v8f;

#define Hdim 128
#define Nn   32768
#define E1n  131072
#define E2n  262144
#define E3n  393216
#define Tn   131072
#define Gn   512

// ------------------------- embeddings --------------------------------------

__global__ void k_embed_nodes(const float* __restrict__ atom_emb,
                              const int* __restrict__ x,
                              float* __restrict__ h0) {
    int v = blockIdx.x, c = threadIdx.x;
    float s = 0.f;
#pragma unroll
    for (int f = 0; f < 9; ++f) {
        int idx = x[v * 9 + f];
        s += atom_emb[((size_t)(f * 128 + idx)) * Hdim + c];
    }
    h0[(size_t)v * Hdim + c] = s;
}

__global__ void k_embed_bonds(const float* __restrict__ bond_emb,
                              const int* __restrict__ ea,
                              float* __restrict__ h1) {
    int e = blockIdx.x, c = threadIdx.x;
    float s = 0.f;
#pragma unroll
    for (int f = 0; f < 3; ++f) {
        int idx = ea[e * 3 + f];
        s += bond_emb[((size_t)(f * 16 + idx)) * Hdim + c];
    }
    h1[(size_t)e * Hdim + c] = s;
}

__global__ void k_edge_pair(const float* __restrict__ h0,
                            const int* __restrict__ ei, int E,
                            float* __restrict__ out) {
    int e = blockIdx.x, c = threadIdx.x;
    int s = ei[e], d = ei[E + e];
    out[(size_t)e * Hdim + c] =
        h0[(size_t)s * Hdim + c] + h0[(size_t)d * Hdim + c];
}

// ------------------------- elementwise helpers -----------------------------

__global__ void k_scale_eps(const float* __restrict__ src,
                            float* __restrict__ dst,
                            const float* __restrict__ epsp, size_t n) {
    float s = 1.0f + *epsp;
    size_t i = (size_t)blockIdx.x * blockDim.x + threadIdx.x;
    size_t stride = (size_t)gridDim.x * blockDim.x;
    for (; i < n; i += stride) dst[i] = s * src[i];
}

__global__ void k_copy(const float* __restrict__ src, float* __restrict__ dst,
                       size_t n) {
    size_t i = (size_t)blockIdx.x * blockDim.x + threadIdx.x;
    size_t stride = (size_t)gridDim.x * blockDim.x;
    for (; i < n; i += stride) dst[i] = src[i];
}

__global__ void k_zero(float* __restrict__ dst, size_t n) {
    size_t i = (size_t)blockIdx.x * blockDim.x + threadIdx.x;
    size_t stride = (size_t)gridDim.x * blockDim.x;
    for (; i < n; i += stride) dst[i] = 0.f;
}

__global__ void k_f32_to_bf16(const float* __restrict__ src,
                              __bf16* __restrict__ dst, size_t n) {
    size_t i = (size_t)blockIdx.x * blockDim.x + threadIdx.x;
    size_t stride = (size_t)gridDim.x * blockDim.x;
    for (; i < n; i += stride) dst[i] = (__bf16)src[i];
}

// stage W^T (row-major [n][k]) in bf16; W is [k][n] row-major f32.
__global__ void k_stage_wt(const float* __restrict__ W,
                           __bf16* __restrict__ Wt) {
    int idx = blockIdx.x * blockDim.x + threadIdx.x; // 16384
    int n = idx >> 7, k = idx & 127;
    Wt[n * Hdim + k] = (__bf16)W[k * Hdim + n];
}

// ------------------------- graph aggregation -------------------------------

__global__ void k_edge_mul_scatter(const float* __restrict__ h0,
                                   const float* __restrict__ hj,
                                   const int* __restrict__ ei, int E,
                                   float* __restrict__ nj,
                                   float* __restrict__ n0) {
    int e = blockIdx.x, c = threadIdx.x;
    int s = ei[e], d = ei[E + e];
    nj[(size_t)e * Hdim + c] +=
        h0[(size_t)s * Hdim + c] * h0[(size_t)d * Hdim + c];
    atomicAdd(&n0[(size_t)s * Hdim + c], hj[(size_t)e * Hdim + c]);
}

__global__ void k_tri(const float* __restrict__ hi,
                      const float* __restrict__ hj,
                      const int* __restrict__ tri, int T,
                      float* __restrict__ nk) {
    int t = blockIdx.x, c = threadIdx.x;
    int a = tri[t], b = tri[T + t], k2 = tri[2 * T + t];
    atomicAdd(&nk[(size_t)k2 * Hdim + c],
              hi[(size_t)a * Hdim + c] * hj[(size_t)b * Hdim + c]);
}

// ------------------------- WMMA GEMM ---------------------------------------
// Z[rows x 128] = A(bf16)[rows x 128] @ W[128 x 128] + bias
// Wt: W^T row-major [n][k] bf16 (so each lane reads 16 contiguous K values
// of its output column). Block = 256 threads = 8 waves; wave owns 16 rows.
__global__ __launch_bounds__(256)
void k_gemm_wmma(const __bf16* __restrict__ A, const __bf16* __restrict__ Wt,
                 const float* __restrict__ bias, float* __restrict__ Z,
                 int rows) {
    __shared__ __bf16 lds_wt[Hdim * Hdim]; // 32 KB
    {
        const uint4* src = (const uint4*)Wt;
        uint4* dst = (uint4*)lds_wt;
        for (int i = threadIdx.x; i < (Hdim * Hdim * 2) / 16; i += 256)
            dst[i] = src[i];
    }
    __syncthreads();

    const int wave = threadIdx.x >> 5;
    const int lane = threadIdx.x & 31;
    const int rowbase = blockIdx.x * 128 + wave * 16;
    if (rowbase >= rows) return; // wave-uniform

    const int half = lane >> 4; // K-half selector (contiguous k = 16*half + s)
    const int l16  = lane & 15; // row (A) / column (B,C,D)

    // 4 A fragments cover K = 0..127 (each WMMA consumes K=32)
    v16bf a[4];
    const __bf16* arow = A + (size_t)(rowbase + l16) * Hdim + half * 16;
#pragma unroll
    for (int kt = 0; kt < 4; ++kt)
        a[kt] = *(const v16bf*)(arow + kt * 32);

#pragma unroll
    for (int nt = 0; nt < 8; ++nt) {
        v8f acc = {};
        const __bf16* bcol = &lds_wt[(nt * 16 + l16) * Hdim + half * 16];
#pragma unroll
        for (int kt = 0; kt < 4; ++kt) {
            v16bf b = *(const v16bf*)(bcol + kt * 32);
            acc = __builtin_amdgcn_wmma_f32_16x16x32_bf16(
                false, a[kt], false, b, (short)0, acc, false, false);
        }
        const float bb = bias ? bias[nt * 16 + l16] : 0.0f;
        const int mbase = rowbase + half * 8; // C/D layout: VGPR r -> M=r(+8)
#pragma unroll
        for (int r = 0; r < 8; ++r)
            Z[(size_t)(mbase + r) * Hdim + nt * 16 + l16] = acc[r] + bb;
    }
}

// ------------------------- batch-norm / ELU / symmetrize -------------------

__global__ void k_colstats(const float* __restrict__ z, int rows,
                           float* __restrict__ stats) {
    int c = threadIdx.x; // 128
    float s = 0.f, s2 = 0.f;
    for (int r = blockIdx.x; r < rows; r += gridDim.x) {
        float v = z[(size_t)r * Hdim + c];
        s += v;
        s2 += v * v;
    }
    atomicAdd(&stats[c], s);
    atomicAdd(&stats[Hdim + c], s2);
}

__global__ void k_bn_elu(const float* __restrict__ z, float* __restrict__ dst,
                         const float* __restrict__ stats,
                         const float* __restrict__ scale,
                         const float* __restrict__ shift, int rows) {
    int c = threadIdx.x;
    float mu = stats[c] / (float)rows;
    float var = stats[Hdim + c] / (float)rows - mu * mu;
    float inv = rsqrtf(var + 1e-5f);
    float sc = scale[c], bi = shift[c];
    for (int r = blockIdx.x; r < rows; r += gridDim.x) {
        float v = sc * (z[(size_t)r * Hdim + c] - mu) * inv + bi;
        dst[(size_t)r * Hdim + c] = v > 0.f ? v : (__expf(v) - 1.0f);
    }
}

__global__ void k_symmetrize(const float* __restrict__ src,
                             const int* __restrict__ invp,
                             float* __restrict__ dst) {
    int e = blockIdx.x, c = threadIdx.x;
    int ie = invp[e];
    dst[(size_t)e * Hdim + c] =
        0.5f * (src[(size_t)e * Hdim + c] + src[(size_t)ie * Hdim + c]);
}

// ------------------------- pooling + MLP -----------------------------------

__global__ void k_pool_scatter(const float* __restrict__ hj,
                               const int* __restrict__ ei,
                               float* __restrict__ nodeAcc) {
    int e = blockIdx.x, c = threadIdx.x;
    int s = ei[e];
    atomicAdd(&nodeAcc[(size_t)s * Hdim + c], hj[(size_t)e * Hdim + c]);
}

__global__ void k_graph_scatter(const float* __restrict__ nodeP,
                                const int* __restrict__ batch0,
                                float* __restrict__ gsum,
                                float* __restrict__ gcnt) {
    int v = blockIdx.x, c = threadIdx.x;
    int b = batch0[v];
    atomicAdd(&gsum[(size_t)b * Hdim + c], nodeP[(size_t)v * Hdim + c]);
    if (c == 0) atomicAdd(&gcnt[b], 1.0f);
}

__global__ void k_mlp(const float* __restrict__ gsum,
                      const float* __restrict__ gcnt,
                      const float* __restrict__ w1, const float* __restrict__ b1,
                      const float* __restrict__ w2, const float* __restrict__ b2,
                      float* __restrict__ out) {
    __shared__ float g[Hdim];
    __shared__ float hid[64];
    int b = blockIdx.x, t = threadIdx.x; // 128 threads
    float cnt = fmaxf(gcnt[b], 1.0f);
    g[t] = gsum[(size_t)b * Hdim + t] / cnt;
    __syncthreads();
    if (t < 64) {
        float s = b1[t];
        for (int c = 0; c < Hdim; ++c) s += g[c] * w1[c * 64 + t];
        hid[t] = s > 0.f ? s : (__expf(s) - 1.0f);
    }
    __syncthreads();
    float s = b2[t];
    for (int j = 0; j < 64; ++j) s += hid[j] * w2[j * Hdim + t];
    out[(size_t)b * Hdim + t] = s;
}

// ------------------------- host orchestration ------------------------------

extern "C" void kernel_launch(void* const* d_in, const int* in_sizes, int n_in,
                              void* d_out, int out_size, void* d_ws,
                              size_t ws_size, hipStream_t stream) {
    const float* atom_emb = (const float*)d_in[0];
    const float* bond_emb = (const float*)d_in[1];
    const float* eps      = (const float*)d_in[2];
    const float* lin_w    = (const float*)d_in[3];
    const float* lin_b    = (const float*)d_in[4];
    const float* nsc      = (const float*)d_in[5];
    const float* nbi      = (const float*)d_in[6];
    const float* pool_w   = (const float*)d_in[7];
    const float* w1       = (const float*)d_in[8];
    const float* b1       = (const float*)d_in[9];
    const float* w2       = (const float*)d_in[10];
    const float* b2       = (const float*)d_in[11];
    const int* x          = (const int*)d_in[12];
    const int* ea         = (const int*)d_in[13];
    const int* ei[3]      = {(const int*)d_in[14], (const int*)d_in[15],
                             (const int*)d_in[16]};
    const int* tri[9]     = {(const int*)d_in[17], (const int*)d_in[18],
                             (const int*)d_in[19], (const int*)d_in[20],
                             (const int*)d_in[21], (const int*)d_in[22],
                             (const int*)d_in[23], (const int*)d_in[24],
                             (const int*)d_in[25]};
    const int* inv[3]     = {(const int*)d_in[26], (const int*)d_in[27],
                             (const int*)d_in[28]};
    const int* batch0     = (const int*)d_in[29];
    float* out            = (float*)d_out;

    const int Ecnt[3] = {E1n, E2n, E3n};
    const int rows[4] = {Nn, E1n, E2n, E3n};
    const size_t off[4] = {0, Nn, (size_t)Nn + E1n, (size_t)Nn + E1n + E2n};
    const size_t ROWS_TOTAL = (size_t)Nn + E1n + E2n + E3n; // 819200
    const int tri_ijk[9][3] = {{1, 1, 1}, {1, 1, 2}, {2, 2, 1},
                               {2, 2, 2}, {1, 2, 3}, {3, 3, 1},
                               {2, 2, 3}, {3, 3, 2}, {3, 3, 3}};

    // workspace carve
    const size_t SB = ROWS_TOTAL * Hdim * sizeof(float); // 419 MB
    char* ws = (char*)d_ws;
    float*  bufH = (float*)ws;                     // h  (all levels)
    float*  bufN = (float*)(ws + SB);              // n / temporaries
    __bf16* bufB = (__bf16*)(ws + 2 * SB);         // bf16 GEMM staging
    __bf16* wt16 = (__bf16*)(ws + 2 * SB + SB / 2);// 32 KB W^T staging
    float*  stats = (float*)(ws + 2 * SB + SB / 2 + 65536); // 256 f
    float*  gsum  = stats + 256;                   // 512 x 128
    float*  gcnt  = gsum + (size_t)Gn * Hdim;      // 512

    // ---- embeddings ----
    k_embed_nodes<<<Nn, Hdim, 0, stream>>>(atom_emb, x, bufH);
    k_embed_bonds<<<E1n, Hdim, 0, stream>>>(bond_emb, ea,
                                            bufH + off[1] * Hdim);
    k_edge_pair<<<E2n, Hdim, 0, stream>>>(bufH, ei[1], E2n,
                                          bufH + off[2] * Hdim);
    k_edge_pair<<<E3n, Hdim, 0, stream>>>(bufH, ei[2], E3n,
                                          bufH + off[3] * Hdim);

    // ---- layers ----
    for (int layer = 0; layer < 2; ++layer) {
        for (int l = 0; l < 4; ++l)
            k_scale_eps<<<2048, 256, 0, stream>>>(
                bufH + off[l] * Hdim, bufN + off[l] * Hdim,
                eps + layer * 4 + l, (size_t)rows[l] * Hdim);

        for (int j = 1; j <= 3; ++j)
            k_edge_mul_scatter<<<Ecnt[j - 1], Hdim, 0, stream>>>(
                bufH, bufH + off[j] * Hdim, ei[j - 1], Ecnt[j - 1],
                bufN + off[j] * Hdim, bufN);

        for (int q = 0; q < 9; ++q) {
            int ti = tri_ijk[q][0], tj = tri_ijk[q][1], tk = tri_ijk[q][2];
            k_tri<<<Tn, Hdim, 0, stream>>>(bufH + off[ti] * Hdim,
                                           bufH + off[tj] * Hdim, tri[q], Tn,
                                           bufN + off[tk] * Hdim);
        }

        // bf16 staging of n, then per-level WMMA GEMM (z overwrites h)
        k_f32_to_bf16<<<4096, 256, 0, stream>>>(bufN, bufB,
                                                ROWS_TOTAL * Hdim);
        for (int l = 0; l < 4; ++l) {
            k_stage_wt<<<64, 256, 0, stream>>>(
                lin_w + ((size_t)layer * 4 + l) * Hdim * Hdim, wt16);
            k_gemm_wmma<<<rows[l] / 128, 256, 0, stream>>>(
                bufB + off[l] * Hdim, wt16,
                lin_b + ((size_t)layer * 4 + l) * Hdim,
                bufH + off[l] * Hdim, rows[l]);
        }

        // batch-norm + ELU (+ inverse-edge symmetrization for edge levels)
        for (int l = 0; l < 4; ++l) {
            const float* sc = nsc + ((size_t)layer * 4 + l) * Hdim;
            const float* bi = nbi + ((size_t)layer * 4 + l) * Hdim;
            k_zero<<<1, 256, 0, stream>>>(stats, 256);
            k_colstats<<<512, Hdim, 0, stream>>>(bufH + off[l] * Hdim,
                                                 rows[l], stats);
            if (l == 0) {
                k_bn_elu<<<1024, Hdim, 0, stream>>>(bufH, bufH, stats, sc, bi,
                                                    rows[l]);
            } else {
                k_bn_elu<<<1024, Hdim, 0, stream>>>(
                    bufH + off[l] * Hdim, bufN + off[l] * Hdim, stats, sc, bi,
                    rows[l]);
                k_symmetrize<<<rows[l], Hdim, 0, stream>>>(
                    bufN + off[l] * Hdim, inv[l - 1], bufH + off[l] * Hdim);
            }
        }
    }

    // ---- pooling: edges -> nodes -> graphs ----
    k_copy<<<2048, 256, 0, stream>>>(bufH, bufN, (size_t)Nn * Hdim);
    for (int j = 1; j <= 3; ++j)
        k_pool_scatter<<<Ecnt[j - 1], Hdim, 0, stream>>>(
            bufH + off[j] * Hdim, ei[j - 1], bufN);

    k_f32_to_bf16<<<2048, 256, 0, stream>>>(bufN, bufB, (size_t)Nn * Hdim);
    k_stage_wt<<<64, 256, 0, stream>>>(pool_w, wt16);
    float* nodeP = bufN + (size_t)Nn * Hdim; // free region
    k_gemm_wmma<<<Nn / 128, 256, 0, stream>>>(bufB, wt16, nullptr, nodeP, Nn);

    k_zero<<<256, 256, 0, stream>>>(gsum, (size_t)Gn * Hdim);
    k_zero<<<1, 256, 0, stream>>>(gcnt, Gn);
    k_graph_scatter<<<Nn, Hdim, 0, stream>>>(nodeP, batch0, gsum, gcnt);
    k_mlp<<<Gn, Hdim, 0, stream>>>(gsum, gcnt, w1, b1, w2, b2, out);
}
